// DeepSphere_4071628997387
// MI455X (gfx1250) — matrix operand order repaired and checked by reading, not compiled, not run on hardware
//
#include <hip/hip_runtime.h>
#include <hip/hip_bf16.h>

typedef __attribute__((ext_vector_type(16))) _Float16 v16h;
typedef __attribute__((ext_vector_type(8)))  float    v8f;

#define BN_EPS 1e-5f
#define NBATCH 32

static inline int cdiv_i(int a, int b) { return (a + b - 1) / b; }

// ---------------- elementwise / graph kernels ----------------

__global__ void k_zero(float* __restrict__ p, int n) {
    int i = blockIdx.x * blockDim.x + threadIdx.x;
    if (i < n) p[i] = 0.0f;
}

__global__ void k_deg(const int* __restrict__ src, const float* __restrict__ w,
                      float* __restrict__ deg, int E) {
    int e = blockIdx.x * blockDim.x + threadIdx.x;
    if (e < E) atomicAdd(deg + src[e], w[e]);
}

__global__ void k_wl(const int* __restrict__ src, const int* __restrict__ dst,
                     const float* __restrict__ w, const float* __restrict__ deg,
                     float* __restrict__ wl, int E) {
    int e = blockIdx.x * blockDim.x + threadIdx.x;
    if (e >= E) return;
    float ds = deg[src[e]], dd = deg[dst[e]];
    float is = ds > 0.0f ? rsqrtf(ds) : 0.0f;
    float id = dd > 0.0f ? rsqrtf(dd) : 0.0f;
    wl[e] = -w[e] * is * id;
}

__global__ void k_scale_copy(const float* __restrict__ in, float* __restrict__ out,
                             float a, int n) {
    int i = blockIdx.x * blockDim.x + threadIdx.x;
    if (i < n) out[i] = a * in[i];
}

// out[b,dst,c] += s * wl[e] * x[b,src,c]   (B = NBATCH batches looped per thread)
__global__ void k_scatter(const float* __restrict__ x, float* __restrict__ out,
                          const int* __restrict__ src, const int* __restrict__ dst,
                          const float* __restrict__ wl, float s, int E, int N, int C) {
    int idx = blockIdx.x * blockDim.x + threadIdx.x;
    if (idx >= E * C) return;
    int e = idx / C, c = idx - e * C;
    float w = s * wl[e];
    size_t so = (size_t)src[e] * C + c;
    size_t dofs = (size_t)dst[e] * C + c;
    size_t stride = (size_t)N * C;
    #pragma unroll 4
    for (int b = 0; b < NBATCH; ++b)
        atomicAdd(out + (size_t)b * stride + dofs, w * x[(size_t)b * stride + so]);
}

// f32 (M x Cin) -> f16 WMMA A-fragment slice for Chebyshev order `korder`.
// Dest layout: Afrag[mt][k][kt][lane][i] (reduction over k,kt contiguous per mt).
//   lane<16:  row = mt*16+lane,    K = kt*32 + {0..7, 16..23}
//   lane>=16: row = mt*16+lane-16, K = kt*32 + {8..15, 24..31}
__global__ void k_cvtA(const float* __restrict__ x, _Float16* __restrict__ A,
                       int Mtiles, int ksteps, int Cin, int korder) {
    int idx = blockIdx.x * blockDim.x + threadIdx.x;
    int total = Mtiles * ksteps * 512;
    if (idx >= total) return;
    int i = idx & 15;
    int lane = (idx >> 4) & 31;
    int kt = (idx >> 9) % ksteps;
    int mt = idx / (ksteps * 512);
    int m = mt * 16 + (lane & 15);
    int khalf = lane >> 4;
    int c = kt * 32 + ((i < 8) ? (khalf * 8 + i) : (16 + khalf * 8 + (i - 8)));
    size_t dstoff = (((size_t)mt * 6 + korder) * ksteps + kt) * 512 + lane * 16 + i;
    A[dstoff] = (c < Cin) ? (_Float16)x[(size_t)m * Cin + c] : (_Float16)0.0f;
}

// W (6,Cin,Cout) f32 -> f16 WMMA B-fragments: Bfrag[nt][k][kt][lane][i]
//   lane holds column n = nt*16+(lane&15); K-rows = kt*32 + {0..7, 16..23} (lane<16), {8..15, 24..31} (lane>=16).
__global__ void k_cvtW(const float* __restrict__ W, _Float16* __restrict__ Wh,
                       int Cin, int Cout, int ksteps) {
    int ntiles = Cout >> 4;
    int total = ntiles * 6 * ksteps * 512;
    int idx = blockIdx.x * blockDim.x + threadIdx.x;
    if (idx >= total) return;
    int i = idx & 15;
    int lane = (idx >> 4) & 31;
    int kt = (idx >> 9) % ksteps;
    int k = (idx / (ksteps * 512)) % 6;
    int nt = idx / (6 * ksteps * 512);
    int n = nt * 16 + (lane & 15);
    int c = kt * 32 + ((i < 8) ? ((lane >> 4) * 8 + i) : (16 + (lane >> 4) * 8 + (i - 8)));
    Wh[idx] = (c < Cin) ? (_Float16)W[((size_t)k * Cin + c) * Cout + n] : (_Float16)0.0f;
}

// HEALPix nested 4->1 average pool: in (B, 4*Nout, C) -> out (B, Nout, C)
__global__ void k_pool4(const float* __restrict__ in, float* __restrict__ out,
                        int Nout, int C) {
    int i = blockIdx.x * blockDim.x + threadIdx.x;
    if (i >= NBATCH * Nout * C) return;
    int c = i % C;
    int n = (i / C) % Nout;
    int b = i / (C * Nout);
    size_t base = ((size_t)b * Nout * 4 + (size_t)n * 4) * C + c;
    out[i] = 0.25f * (in[base] + in[base + C] + in[base + 2 * (size_t)C] + in[base + 3 * (size_t)C]);
}

__global__ void k_gmean(const float* __restrict__ in, float* __restrict__ out,
                        int N, int C) {
    int i = blockIdx.x * blockDim.x + threadIdx.x;
    if (i >= NBATCH * C) return;
    int b = i / C, c = i % C;
    float s = 0.0f;
    for (int n = 0; n < N; ++n) s += in[((size_t)b * N + n) * C + c];
    out[i] = s / (float)N;
}

__global__ void k_fc(const float* __restrict__ in, const float* __restrict__ W,
                     const float* __restrict__ bias, float* __restrict__ out,
                     int Cin, int Cout, int relu) {
    int i = blockIdx.x * blockDim.x + threadIdx.x;
    if (i >= NBATCH * Cout) return;
    int b = i / Cout, o = i % Cout;
    float s = bias[o];
    for (int c = 0; c < Cin; ++c) s += in[(size_t)b * Cin + c] * W[(size_t)c * Cout + o];
    out[i] = relu ? fmaxf(s, 0.0f) : s;
}

// ---------------- fused WMMA GEMM + bias + BN + ReLU ----------------
// One launch per ChebConv: reduces over all 6 orders (K_T = 6*Kpad/32 WMMA steps),
// accumulators stay in VGPRs the whole time, epilogue applies bias/BN/ReLU.
// 8 waves/block; each wave computes FOUR 16x16 tiles (64 rows) for one N-tile.
// Software-pipelined (depth 1): fragments for step kt+1 are fetched before the
// 4 WMMAs of step kt issue, so loads overlap the matrix pipe.
// M % 512 == 0, Cout % 16 == 0 -> no branches around WMMA, EXEC all-ones.
template <int KT>
__global__ void k_gemm_fused(const _Float16* __restrict__ Afr, const _Float16* __restrict__ Bfr,
                             const float* __restrict__ bc, const float* __restrict__ g,
                             const float* __restrict__ be, float* __restrict__ hout, int Ncols) {
    const int wave = threadIdx.x >> 5;
    const int lane = threadIdx.x & 31;
    const int half_ = lane >> 4;
    const int l16 = lane & 15;
    const int mt0 = (blockIdx.x * 8 + wave) * 4;  // 4 consecutive M-tiles per wave
    const int nt = blockIdx.y;
    const int n0 = nt * 16;

    const _Float16* ab = Afr + (size_t)mt0 * KT * 512 + lane * 16;
    const _Float16* bb = Bfr + (size_t)nt * KT * 512 + lane * 16;
    const size_t sA = (size_t)KT * 512;  // A stride between M-tiles

    v8f acc[4];
    #pragma unroll
    for (int t = 0; t < 4; ++t)
        #pragma unroll
        for (int r = 0; r < 8; ++r) acc[t][r] = 0.0f;

    v16h bcur = *(const v16h*)(bb);
    v16h ac0 = *(const v16h*)(ab);
    v16h ac1 = *(const v16h*)(ab + sA);
    v16h ac2 = *(const v16h*)(ab + 2 * sA);
    v16h ac3 = *(const v16h*)(ab + 3 * sA);

    #pragma unroll
    for (int kt = 0; kt < KT; ++kt) {
        v16h bnx, an0, an1, an2, an3;
        if (kt + 1 < KT) {
            const int o = (kt + 1) * 512;
            bnx = *(const v16h*)(bb + o);
            an0 = *(const v16h*)(ab + o);
            an1 = *(const v16h*)(ab + sA + o);
            an2 = *(const v16h*)(ab + 2 * sA + o);
            an3 = *(const v16h*)(ab + 3 * sA + o);
        }
        acc[0] = __builtin_amdgcn_wmma_f32_16x16x32_f16(false, ac0, false, bcur, (short)0, acc[0], false, false);
        acc[1] = __builtin_amdgcn_wmma_f32_16x16x32_f16(false, ac1, false, bcur, (short)0, acc[1], false, false);
        acc[2] = __builtin_amdgcn_wmma_f32_16x16x32_f16(false, ac2, false, bcur, (short)0, acc[2], false, false);
        acc[3] = __builtin_amdgcn_wmma_f32_16x16x32_f16(false, ac3, false, bcur, (short)0, acc[3], false, false);
        if (kt + 1 < KT) { bcur = bnx; ac0 = an0; ac1 = an1; ac2 = an2; ac3 = an3; }
    }

    // epilogue: (acc + bias) * gamma/sqrt(1+eps) + beta, ReLU
    const int col = n0 + l16;
    const float bcv = bc[col];
    const float gs = g[col] * rsqrtf(1.0f + BN_EPS);
    const float bev = be[col];
    #pragma unroll
    for (int t = 0; t < 4; ++t)
        #pragma unroll
        for (int r = 0; r < 8; ++r) {
            float v = (acc[t][r] + bcv) * gs + bev;
            hout[(size_t)((mt0 + t) * 16 + half_ * 8 + r) * Ncols + col] = fmaxf(v, 0.0f);
        }
}

// ---------------- host orchestration ----------------

static void cheb_block(hipStream_t st, const float* h, float* hout,
                       float* Ts0, float* Ts1, float* Ts2,
                       _Float16* Af16, _Float16* Wh,
                       const int* src, const int* dst, const float* wl, int E, int N,
                       int Cin, int Cout, const float* Wc, const float* bc,
                       const float* g, const float* be) {
    float* Ts[3] = {Ts0, Ts1, Ts2};
    const int M = NBATCH * N;
    const int Kpad = Cin < 32 ? 32 : Cin;
    const int ksteps = Kpad >> 5;
    const int Mtiles = M >> 4;
    const int ntiles = Cout >> 4;
    const int KT = 6 * ksteps;

    { int n = ntiles * 6 * ksteps * 512; k_cvtW<<<cdiv_i(n, 256), 256, 0, st>>>(Wc, Wh, Cin, Cout, ksteps); }

    auto convA = [&](const float* t, int k) {
        int n = Mtiles * ksteps * 512;
        k_cvtA<<<cdiv_i(n, 256), 256, 0, st>>>(t, Af16, Mtiles, ksteps, Cin, k);
    };

    // Chebyshev recursion: build all 6 order slices of the A-fragment buffer
    convA(h, 0);
    k_zero<<<cdiv_i(M * Cin, 256), 256, 0, st>>>(Ts[0], M * Cin);
    k_scatter<<<cdiv_i(E * Cin, 256), 256, 0, st>>>(h, Ts[0], src, dst, wl, 1.0f, E, N, Cin);
    convA(Ts[0], 1);
    const float* Tm2 = h;
    float* Tm1 = Ts[0];
    int slot = 1;
    for (int k = 2; k < 6; ++k) {
        float* Tn = Ts[slot];
        k_scale_copy<<<cdiv_i(M * Cin, 256), 256, 0, st>>>(Tm2, Tn, -1.0f, M * Cin);
        k_scatter<<<cdiv_i(E * Cin, 256), 256, 0, st>>>(Tm1, Tn, src, dst, wl, 2.0f, E, N, Cin);
        convA(Tn, k);
        Tm2 = Tm1; Tm1 = Tn; slot = (slot + 1) % 3;
    }

    // one fused GEMM over the whole 6*Kpad reduction + bias/BN/ReLU epilogue
    dim3 ggrid(M / 512, ntiles);
    switch (KT) {
        case 6:  k_gemm_fused<6><<<ggrid, 256, 0, st>>>(Af16, Wh, bc, g, be, hout, Cout); break;
        case 12: k_gemm_fused<12><<<ggrid, 256, 0, st>>>(Af16, Wh, bc, g, be, hout, Cout); break;
        default: k_gemm_fused<24><<<ggrid, 256, 0, st>>>(Af16, Wh, bc, g, be, hout, Cout); break;
    }
}

extern "C" void kernel_launch(void* const* d_in, const int* in_sizes, int n_in,
                              void* d_out, int out_size, void* d_ws, size_t ws_size,
                              hipStream_t stream) {
    const float* x     = (const float*)d_in[0];
    const int*   src64 = (const int*)d_in[1];
    const int*   dst64 = (const int*)d_in[2];
    const float* w64   = (const float*)d_in[3];
    const int*   src32 = (const int*)d_in[4];
    const int*   dst32 = (const int*)d_in[5];
    const float* w32   = (const float*)d_in[6];
    const int*   src16 = (const int*)d_in[7];
    const int*   dst16 = (const int*)d_in[8];
    const float* w16   = (const float*)d_in[9];
    const float* Wc[7]; const float* bc[7]; const float* gg[7]; const float* bb[7];
    for (int i = 0; i < 7; ++i) {
        Wc[i] = (const float*)d_in[10 + 4 * i];
        bc[i] = (const float*)d_in[11 + 4 * i];
        gg[i] = (const float*)d_in[12 + 4 * i];
        bb[i] = (const float*)d_in[13 + 4 * i];
    }
    const float* Wf1 = (const float*)d_in[38];
    const float* bf1 = (const float*)d_in[39];
    const float* Wf2 = (const float*)d_in[40];
    const float* bf2 = (const float*)d_in[41];
    const float* Wo  = (const float*)d_in[42];
    const float* bo  = (const float*)d_in[43];
    float* out = (float*)d_out;

    const int E64 = in_sizes[1], E32 = in_sizes[4], E16 = in_sizes[7];

    // workspace carving
    char* p = (char*)d_ws;
    auto carve = [&](size_t bytes) -> void* {
        void* r = (void*)p;
        p += (bytes + 255) & ~(size_t)255;
        return r;
    };
    const size_t MAXH = (size_t)NBATCH * 12288 * 64;   // activations: 25.2M floats
    const size_t MAXT = (size_t)NBATCH * 12288 * 32;   // Chebyshev scratch: 12.6M floats
    const size_t MAXA = (size_t)NBATCH * 12288 * 192;  // A-fragments (M x 6*Kpad f16)
    float*    hA   = (float*)carve(MAXH * 4);
    float*    T0   = (float*)carve(MAXT * 4);
    float*    T1   = (float*)carve(MAXT * 4);
    float*    T2   = (float*)carve(MAXT * 4);
    _Float16* Af16 = (_Float16*)carve(MAXA * 2);
    _Float16* Wh   = (_Float16*)carve((size_t)6 * 128 * 256 * 2);
    float*    deg  = (float*)carve(12288 * 4);
    float*    wl64 = (float*)carve((size_t)E64 * 4);
    float*    wl32 = (float*)carve((size_t)E32 * 4);
    float*    wl16 = (float*)carve((size_t)E16 * 4);
    float*    feat = (float*)carve(NBATCH * 256 * 4);
    float*    z1   = (float*)carve(NBATCH * 256 * 4);
    float*    z2   = (float*)carve(NBATCH * 128 * 4);

    // scaled-Laplacian edge weights per level
    auto lap = [&](const int* s, const int* d, const float* w, float* wl, int E, int N) {
        k_zero<<<cdiv_i(N, 256), 256, 0, stream>>>(deg, N);
        k_deg<<<cdiv_i(E, 256), 256, 0, stream>>>(s, w, deg, E);
        k_wl<<<cdiv_i(E, 256), 256, 0, stream>>>(s, d, w, deg, wl, E);
    };
    lap(src64, dst64, w64, wl64, E64, 12288);
    lap(src32, dst32, w32, wl32, E32, 3072);
    lap(src16, dst16, w16, wl16, E16, 768);

    // x is (B,1,12288); transposed (B,N,1) has identical layout since C==1.
    cheb_block(stream, x,  hA, T0, T1, T2, Af16, Wh, src64, dst64, wl64, E64, 12288, 1,  32, Wc[0], bc[0], gg[0], bb[0]);
    cheb_block(stream, hA, hA, T0, T1, T2, Af16, Wh, src64, dst64, wl64, E64, 12288, 32, 32, Wc[1], bc[1], gg[1], bb[1]);
    cheb_block(stream, hA, hA, T0, T1, T2, Af16, Wh, src64, dst64, wl64, E64, 12288, 32, 64, Wc[2], bc[2], gg[2], bb[2]);

    k_pool4<<<cdiv_i(NBATCH * 3072 * 64, 256), 256, 0, stream>>>(hA, T0, 3072, 64);
    hipMemcpyAsync(hA, T0, (size_t)NBATCH * 3072 * 64 * 4, hipMemcpyDeviceToDevice, stream);

    cheb_block(stream, hA, hA, T0, T1, T2, Af16, Wh, src32, dst32, wl32, E32, 3072, 64, 64,  Wc[3], bc[3], gg[3], bb[3]);
    cheb_block(stream, hA, hA, T0, T1, T2, Af16, Wh, src32, dst32, wl32, E32, 3072, 64, 128, Wc[4], bc[4], gg[4], bb[4]);

    k_pool4<<<cdiv_i(NBATCH * 768 * 128, 256), 256, 0, stream>>>(hA, T0, 768, 128);
    hipMemcpyAsync(hA, T0, (size_t)NBATCH * 768 * 128 * 4, hipMemcpyDeviceToDevice, stream);

    cheb_block(stream, hA, hA, T0, T1, T2, Af16, Wh, src16, dst16, wl16, E16, 768, 128, 128, Wc[5], bc[5], gg[5], bb[5]);
    cheb_block(stream, hA, hA, T0, T1, T2, Af16, Wh, src16, dst16, wl16, E16, 768, 128, 256, Wc[6], bc[6], gg[6], bb[6]);

    k_pool4<<<cdiv_i(NBATCH * 192 * 256, 256), 256, 0, stream>>>(hA, T0, 192, 256);
    hipMemcpyAsync(hA, T0, (size_t)NBATCH * 192 * 256 * 4, hipMemcpyDeviceToDevice, stream);

    // head: global mean pool + MLP
    k_gmean<<<cdiv_i(NBATCH * 256, 256), 256, 0, stream>>>(hA, feat, 192, 256);
    k_fc<<<cdiv_i(NBATCH * 256, 256), 256, 0, stream>>>(feat, Wf1, bf1, z1, 256, 256, 1);
    k_fc<<<cdiv_i(NBATCH * 128, 256), 256, 0, stream>>>(z1, Wf2, bf2, z2, 256, 128, 1);
    k_fc<<<cdiv_i(NBATCH * 1, 256), 256, 0, stream>>>(z2, Wo, bo, out, 128, 1, 0);
}
